// MTREncoder_56659208569436
// MI455X (gfx1250) — matrix operand order repaired and checked by reading, not compiled, not run on hardware
//
#include <hip/hip_runtime.h>
#include <hip/hip_bf16.h>

typedef _Float16 h16;
typedef __attribute__((ext_vector_type(16))) _Float16 v16h;
typedef __attribute__((ext_vector_type(8)))  _Float16 v8h;
typedef __attribute__((ext_vector_type(8)))  float    v8f;

#define DEVFN __device__ __forceinline__

constexpr int Bn   = 8;
constexpr int No   = 64;
constexpr int Tt   = 12;
constexpr int Pp   = 768;
constexpr int NPt  = 20;
constexpr int Dm   = 256;
constexpr int HD   = 32;
constexpr int KNN  = 16;
constexpr int NT   = No * Tt;        // 768 obj tokens / batch
constexpr int NTOK = NT + Pp;        // 1536 tokens / batch
constexpr int R_TOK = Bn * NTOK;     // 12288
constexpr int R_OBJ = Bn * No * Tt;  // 6144
constexpr int R_MP  = Bn * Pp * NPt; // 122880
constexpr int R_MAP = Bn * Pp;       // 6144

DEVFN v8f wmma_step(v16h a, v16h b, v8f c)
{
    return __builtin_amdgcn_wmma_f32_16x16x32_f16(false, a, false, b,
                                                  (short)0, c, false, false);
}

// ---------------------------------------------------------------------------
// Generic f16 WMMA GEMM:  out = act( (A[M,Kp] @ Wt[N,Kp]^T + bias) * alpha )
// Wt is the transposed weight (row n = column n of W), f16, K padded to 32.
// One wave computes a 16x64 strip: 1 A-fragment reused against 4 B-fragments
// per 32-K step (4x v_wmma_f32_16x16x32_f16), 13 FLOP/B fragment traffic.
// N must be a multiple of 64; M a multiple of 16.
// ---------------------------------------------------------------------------
__global__ void k_gemm(const h16* __restrict__ A, const h16* __restrict__ Bt,
                       const float* __restrict__ bias, float alpha, int relu,
                       float* __restrict__ outF, h16* __restrict__ outH,
                       int M, int N, int Kp, int strips)
{
    int lane  = threadIdx.x & 31;
    int wave  = threadIdx.x >> 5;
    int strip = blockIdx.x * 8 + wave;
    if (strip >= strips) return;               // wave-uniform: EXEC stays full
    int stripsN = N >> 6;
    int tm = strip / stripsN;
    int sn = strip - tm * stripsN;
    int r  = lane & 15;
    int hh = lane >> 4;

    const h16* ap = A  + (size_t)((tm << 4) + r) * Kp + 16 * hh;
    const h16* bp = Bt + (size_t)((sn << 6) + r) * Kp + 16 * hh;
    const size_t bstep = (size_t)16 * Kp;      // next 16-col tile of Bt

    v8f acc0 = {}, acc1 = {}, acc2 = {}, acc3 = {};
    for (int kb = 0; kb < Kp; kb += 32) {
        v16h a  = *(const v16h*)(ap + kb);         // K = kb+16h..+15
        v16h b0 = *(const v16h*)(bp + kb);                 // K = kb+16h..+15
        v16h b1 = *(const v16h*)(bp + bstep     + kb);
        v16h b2 = *(const v16h*)(bp + bstep * 2 + kb);
        v16h b3 = *(const v16h*)(bp + bstep * 3 + kb);
        acc0 = wmma_step(a, b0, acc0);
        acc1 = wmma_step(a, b1, acc1);
        acc2 = wmma_step(a, b2, acc2);
        acc3 = wmma_step(a, b3, acc3);
    }

#pragma unroll
    for (int t = 0; t < 4; ++t) {
        v8f acc = (t == 0) ? acc0 : (t == 1) ? acc1 : (t == 2) ? acc2 : acc3;
        int ncol = (sn << 6) + 16 * t + r;
        float bs = bias ? bias[ncol] : 0.0f;
#pragma unroll
        for (int i = 0; i < 8; ++i) {
            int mrow = (tm << 4) + 8 * hh + i;
            float v = (acc[i] + bs) * alpha;
            if (relu) v = v > 0.0f ? v : 0.0f;
            size_t o = (size_t)mrow * N + ncol;
            if (outF) outF[o] = v;
            if (outH) outH[o] = (h16)v;
        }
    }
}

// Transpose + convert weight W[K,N] (f32) -> Wt[N,Kp] (f16), zero pad K..Kp.
__global__ void k_wt(const float* __restrict__ W, h16* __restrict__ Wt,
                     int K, int N, int Kp)
{
    int i = blockIdx.x * 256 + threadIdx.x;
    if (i >= N * Kp) return;
    int n = i / Kp, k = i - n * Kp;
    Wt[i] = (k < K) ? (h16)W[(size_t)k * N + n] : (h16)0.0f;
}

// out16 = f16(a (+ b))
__global__ void k_cast16(const float* __restrict__ a, const float* __restrict__ b,
                         h16* __restrict__ o, int n)
{
    int i = blockIdx.x * 256 + threadIdx.x;
    if (i >= n) return;
    float v = a[i];
    if (b) v += b[i];
    o[i] = (h16)v;
}

// zero rows whose mask is 0
__global__ void k_maskrows(float* __restrict__ x, const int* __restrict__ m,
                           int R, int C)
{
    int i = blockIdx.x * 256 + threadIdx.x;
    if (i >= R * C) return;
    if (!m[i / C]) x[i] = 0.0f;
}

// out[g,c] = max_l in[(g*L+l), c]
__global__ void k_poolmax(const float* __restrict__ in, float* __restrict__ out,
                          int G, int L, int C)
{
    int i = blockIdx.x * 256 + threadIdx.x;
    if (i >= G * C) return;
    int g = i / C, c = i - g * C;
    const float* p = in + (size_t)g * L * C + c;
    float mx = p[0];
    for (int l = 1; l < L; ++l) mx = fmaxf(mx, p[(size_t)l * C]);
    out[i] = mx;
}

// o16[r, 0:C1]=a[r]; o16[r, C1:C1+C2]=pool[r/L]  (f16)
__global__ void k_concat16(const float* __restrict__ a, const float* __restrict__ pool,
                           h16* __restrict__ o, int R, int C1, int C2, int L)
{
    int C = C1 + C2;
    int i = blockIdx.x * 256 + threadIdx.x;
    if (i >= R * C) return;
    int r = i / C, c = i - r * C;
    float v = (c < C1) ? a[(size_t)r * C1 + c]
                       : pool[(size_t)(r / L) * C2 + (c - C1)];
    o[i] = (h16)v;
}

// obj_in f16 [R_OBJ,32]: 29 traj ch + mask + 2 pad
__global__ void k_obj_in(const float* __restrict__ trj, const int* __restrict__ msk,
                         h16* __restrict__ o)
{
    int i = blockIdx.x * 256 + threadIdx.x;
    if (i >= R_OBJ * 32) return;
    int row = i >> 5, c = i & 31;
    float v = 0.0f;
    if (c < 29)       v = trj[(size_t)row * 29 + c];
    else if (c == 29) v = (msk[row] != 0) ? 1.0f : 0.0f;
    o[i] = (h16)v;
}

// map points f16 [R_MP,32]: 9 ch + pad
__global__ void k_map_in(const float* __restrict__ mp, h16* __restrict__ o)
{
    int i = blockIdx.x * 256 + threadIdx.x;
    if (i >= R_MP * 32) return;
    int row = i >> 5, c = i & 31;
    o[i] = (c < 9) ? (h16)mp[(size_t)row * 9 + c] : (h16)0.0f;
}

// token mask + positions
__global__ void k_tok_meta(const int* __restrict__ omask, const int* __restrict__ mmask,
                           const float* __restrict__ opos, const float* __restrict__ mcen,
                           int* __restrict__ tmask, float* __restrict__ tpos)
{
    int t = blockIdx.x * 256 + threadIdx.x;
    if (t >= R_TOK) return;
    int b = t / NTOK, r = t - b * NTOK;
    int v; float x, y, z;
    if (r < NT) {
        int row = b * NT + r;
        v = (omask[row] != 0);
        x = opos[row * 3 + 0]; y = opos[row * 3 + 1]; z = opos[row * 3 + 2];
    } else {
        int p = r - NT, base = b * Pp + p;
        v = 0;
        for (int j = 0; j < NPt; ++j) v |= (mmask[base * NPt + j] != 0);
        x = mcen[base * 3 + 0]; y = mcen[base * 3 + 1]; z = mcen[base * 3 + 2];
    }
    tmask[t] = v;
    tpos[t * 3 + 0] = x; tpos[t * 3 + 1] = y; tpos[t * 3 + 2] = z;
}

// X[tok,c] = (obj|map feature) * tmask
__global__ void k_tokens(const float* __restrict__ objf, const float* __restrict__ mapf,
                         const int* __restrict__ tmask, float* __restrict__ X)
{
    int i = blockIdx.x * 256 + threadIdx.x;
    if (i >= R_TOK * Dm) return;
    int tok = i / Dm, c = i - tok * Dm;
    int b = tok / NTOK, r = tok - b * NTOK;
    float v = (r < NT) ? objf[(size_t)(b * NT + r) * Dm + c]
                       : mapf[(size_t)(b * Pp + (r - NT)) * Dm + c];
    X[i] = tmask[tok] ? v : 0.0f;
}

// kNN: 1 block per scene, key set staged in LDS, 16-entry insertion sort
__global__ void k_knn(const float* __restrict__ tpos, const int* __restrict__ tmask,
                      int* __restrict__ idx, int* __restrict__ nbv)
{
    __shared__ float sx[NTOK], sy[NTOK], sz[NTOK], sq[NTOK];
    __shared__ int   sv[NTOK];
    __shared__ float dls[256 * KNN];
    __shared__ int   ils[256 * KNN];
    int b = blockIdx.x;
    for (int j = threadIdx.x; j < NTOK; j += 256) {
        int t = b * NTOK + j;
        float x = tpos[t * 3], y = tpos[t * 3 + 1], z = tpos[t * 3 + 2];
        sx[j] = x; sy[j] = y; sz[j] = z;
        sq[j] = x * x + y * y + z * z;
        sv[j] = tmask[t];
    }
    __syncthreads();
    float* d  = dls + threadIdx.x * KNN;
    int*   id = ils + threadIdx.x * KNN;
    for (int q = threadIdx.x; q < NTOK; q += 256) {
        float qx = sx[q], qy = sy[q], qz = sz[q], qs = sq[q];
        for (int k = 0; k < KNN; ++k) { d[k] = 3.0e38f; id[k] = 0; }
        for (int j = 0; j < NTOK; ++j) {
            if (!sv[j]) continue;
            float dj = qs + sq[j] - 2.0f * (qx * sx[j] + qy * sy[j] + qz * sz[j]);
            if (dj < d[KNN - 1]) {
                int p = KNN - 1;
                while (p > 0 && dj < d[p - 1]) { d[p] = d[p - 1]; id[p] = id[p - 1]; --p; }
                d[p] = dj; id[p] = j;
            }
        }
        int base = (b * NTOK + q) * KNN;
        for (int k = 0; k < KNN; ++k) {
            idx[base + k] = id[k];
            nbv[base + k] = (d[k] < 1.0e30f) ? 1 : 0;
        }
    }
}

// sine positional embedding, one thread per (token, pair)
__global__ void k_posemb(const float* __restrict__ tpos, float* __restrict__ pe)
{
    int i = blockIdx.x * 256 + threadIdx.x;
    if (i >= R_TOK * 128) return;
    int tok = i >> 7, p = i & 127;
    int iloc = p & 63;
    float v = (p < 64) ? tpos[tok * 3 + 1] : tpos[tok * 3 + 0]; // y first, then x
    float freq = powf(10000.0f, (float)iloc * (1.0f / 64.0f));
    float arg = v * 6.283185307179586f / freq;
    pe[(size_t)tok * Dm + 2 * p + 0] = sinf(arg);
    pe[(size_t)tok * Dm + 2 * p + 1] = cosf(arg);
}

DEVFN float wred32(float v)
{
    for (int o = 16; o > 0; o >>= 1) v += __shfl_xor(v, o, 32);
    return v;
}

// local attention: one wave per (token, head); lane = head-dim element
__global__ void k_attn(const float* __restrict__ Q, const float* __restrict__ K,
                       const float* __restrict__ V, const int* __restrict__ idx,
                       const int* __restrict__ nbv, float* __restrict__ ctx)
{
    int tok  = blockIdx.x;
    int h    = threadIdx.x >> 5;
    int lane = threadIdx.x & 31;
    int b    = tok / NTOK;
    float qd = Q[(size_t)tok * Dm + h * HD + lane];

    float lg[KNN]; int jj[KNN];
#pragma unroll
    for (int k = 0; k < KNN; ++k) {
        int j = idx[tok * KNN + k];
        jj[k] = j;
        float kd = K[(size_t)(b * NTOK + j) * Dm + h * HD + lane];
        float p = wred32(qd * kd);
        lg[k] = nbv[tok * KNN + k] ? p : -1.0e9f;
    }
    float mx = lg[0];
#pragma unroll
    for (int k = 1; k < KNN; ++k) mx = fmaxf(mx, lg[k]);
    float e[KNN]; float se = 0.0f;
#pragma unroll
    for (int k = 0; k < KNN; ++k) { e[k] = expf(lg[k] - mx); se += e[k]; }
    float inv = 1.0f / se;
    float acc = 0.0f;
#pragma unroll
    for (int k = 0; k < KNN; ++k)
        acc += e[k] * inv * V[(size_t)(b * NTOK + jj[k]) * Dm + h * HD + lane];
    ctx[(size_t)tok * Dm + h * HD + lane] = acc;
}

// out = LN(src + delta) * g + be ; one wave per 256-wide row; optional f16 copy
__global__ void k_ln(const float* __restrict__ src, const float* __restrict__ delta,
                     const float* __restrict__ g, const float* __restrict__ be,
                     float* __restrict__ outF, h16* __restrict__ outH)
{
    int row  = blockIdx.x * 8 + (threadIdx.x >> 5);
    int lane = threadIdx.x & 31;
    float v[8]; float s = 0.0f;
#pragma unroll
    for (int j = 0; j < 8; ++j) {
        size_t o = (size_t)row * Dm + j * 32 + lane;
        v[j] = src[o] + delta[o];
        s += v[j];
    }
    float mean = wred32(s) * (1.0f / 256.0f);
    float q = 0.0f;
#pragma unroll
    for (int j = 0; j < 8; ++j) { float dv = v[j] - mean; q += dv * dv; }
    float rstd = rsqrtf(wred32(q) * (1.0f / 256.0f) + 1e-5f);
#pragma unroll
    for (int j = 0; j < 8; ++j) {
        int c = j * 32 + lane;
        float o = (v[j] - mean) * rstd * g[c] + be[c];
        size_t off = (size_t)row * Dm + c;
        outF[off] = o;
        if (outH) outH[off] = (h16)o;
    }
}

// final [B, No+P, T, D] assembly with token masking + map broadcast over T
__global__ void k_out(const float* __restrict__ X, const int* __restrict__ tmask,
                      float* __restrict__ out, int n)
{
    int i = blockIdx.x * 256 + threadIdx.x;
    if (i >= n) return;
    int c = i & (Dm - 1);
    int t = (i / Dm) % Tt;
    int a = (i / (Dm * Tt)) % (No + Pp);
    int b = i / (Dm * Tt * (No + Pp));
    int tok = (a < No) ? (b * NTOK + a * Tt + t) : (b * NTOK + NT + (a - No));
    out[i] = tmask[tok] ? X[(size_t)tok * Dm + c] : 0.0f;
}

// ---------------------------------------------------------------------------
// Host side
// ---------------------------------------------------------------------------
static inline int gdiv(int n) { return (n + 255) / 256; }

struct Bump {
    char* base; size_t off;
    void* take(size_t bytes) {
        off = (off + 255) & ~(size_t)255;
        void* p = base + off; off += bytes; return p;
    }
    float* f32(size_t n) { return (float*)take(n * 4); }
    h16*   f16(size_t n) { return (h16*)take(n * 2); }
    int*   i32(size_t n) { return (int*)take(n * 4); }
};

static void gemm(hipStream_t st, const h16* A, const h16* Bt, const float* bias,
                 float alpha, int relu, float* oF, h16* oH, int M, int N, int Kp)
{
    int strips = (M / 16) * (N / 64);
    int blocks = (strips + 7) / 8;
    k_gemm<<<blocks, 256, 0, st>>>(A, Bt, bias, alpha, relu, oF, oH, M, N, Kp, strips);
}

extern "C" void kernel_launch(void* const* d_in, const int* in_sizes, int n_in,
                              void* d_out, int out_size, void* d_ws, size_t ws_size,
                              hipStream_t stream)
{
    // --- canonical slot table (sizes) ---
    // 0..5 inputs, 6..12 agent, 13..21 map, 22+16l layers
    static const long long slotSize[118] = {
        178176, 6144, 1105920, 122880, 18432, 18432,
        7680, 131072, 65536, 65536, 256, 65536, 256,
        576, 4096, 4096, 8192, 4096, 4096, 64, 16384, 256,
        // 6 layers x {wq,bq,wk,bk,wv,bv,wo,bo,g1,be1,g2,be2,w1,fb1,w2,fb2}
        65536,256,65536,256,65536,256,65536,256,256,256,256,256,262144,1024,262144,256,
        65536,256,65536,256,65536,256,65536,256,256,256,256,256,262144,1024,262144,256,
        65536,256,65536,256,65536,256,65536,256,256,256,256,256,262144,1024,262144,256,
        65536,256,65536,256,65536,256,65536,256,256,256,256,256,262144,1024,262144,256,
        65536,256,65536,256,65536,256,65536,256,256,256,256,256,262144,1024,262144,256,
        65536,256,65536,256,65536,256,65536,256,256,256,256,256,262144,1024,262144,256
    };
    int ordA[118], ordB[118];
    for (int i = 0; i < 118; ++i) ordA[i] = i;
    {   // jax tree_flatten (alphabetical dict keys) candidate
        int k = 0;
        ordB[k++] = 2; ordB[k++] = 5; ordB[k++] = 3;   // map_pl, map_cen, map_mask
        ordB[k++] = 0; ordB[k++] = 1; ordB[k++] = 4;   // obj_trajs, obj_mask, obj_pos
        const int ag[7] = {7, 8, 10, 12, 9, 11, 6};    // mlp0,mlp1,ob0,ob1,ow0,ow1,pre
        for (int i = 0; i < 7; ++i) ordB[k++] = ag[i];
        const int ll[16] = {9, 11, 3, 7, 1, 5, 13, 15, 8, 10, 12, 14, 2, 6, 0, 4};
        for (int l = 0; l < 6; ++l)
            for (int i = 0; i < 16; ++i) ordB[k++] = 22 + 16 * l + ll[i];
        const int mg[9] = {16, 17, 19, 21, 18, 20, 13, 14, 15}; // mlp0,mlp1,ob0,ob1,ow0,ow1,pre0,pre1,pre2
        for (int i = 0; i < 9; ++i) ordB[k++] = mg[i];
    }
    const int* ord = ordA;
    if (n_in == 118) {
        bool okA = true, okB = true;
        for (int i = 0; i < 118; ++i) {
            okA = okA && ((long long)in_sizes[i] == slotSize[ordA[i]]);
            okB = okB && ((long long)in_sizes[i] == slotSize[ordB[i]]);
        }
        if (!okA && okB) ord = ordB;
    }
    const void* slot[118];
    int lim = (n_in < 118) ? n_in : 118;
    for (int i = 0; i < 118; ++i) slot[i] = nullptr;
    for (int i = 0; i < lim; ++i) slot[ord[i]] = d_in[i];

    const float* obj_trajs = (const float*)slot[0];
    const int*   omask     = (const int*)slot[1];
    const float* map_pl    = (const float*)slot[2];
    const int*   mmask     = (const int*)slot[3];
    const float* opos      = (const float*)slot[4];
    const float* mcen      = (const float*)slot[5];
    const float *a_pre = (const float*)slot[6],  *a_m0 = (const float*)slot[7],
                *a_m1  = (const float*)slot[8],  *a_ow0 = (const float*)slot[9],
                *a_ob0 = (const float*)slot[10], *a_ow1 = (const float*)slot[11],
                *a_ob1 = (const float*)slot[12];
    const float *m_p0 = (const float*)slot[13], *m_p1 = (const float*)slot[14],
                *m_p2 = (const float*)slot[15], *m_m0 = (const float*)slot[16],
                *m_m1 = (const float*)slot[17], *m_ow0 = (const float*)slot[18],
                *m_ob0 = (const float*)slot[19], *m_ow1 = (const float*)slot[20],
                *m_ob1 = (const float*)slot[21];

    // --- workspace layout ---
    Bump ws{(char*)d_ws, 0};
    h16* wt_apre = ws.f16(256 * 32);
    h16* wt_am0  = ws.f16(256 * 512);
    h16* wt_am1  = ws.f16(256 * 256);
    h16* wt_aw0  = ws.f16(256 * 256);
    h16* wt_aw1  = ws.f16(256 * 256);
    h16* wt_mp0  = ws.f16(64 * 32);
    h16* wt_mp1  = ws.f16(64 * 64);
    h16* wt_mp2  = ws.f16(64 * 64);
    h16* wt_mm0  = ws.f16(64 * 128);
    h16* wt_mm1  = ws.f16(64 * 64);
    h16* wt_mw0  = ws.f16(64 * 64);
    h16* wt_mw1  = ws.f16(256 * 64);
    h16 *wt_q[6], *wt_k[6], *wt_v[6], *wt_o[6], *wt_1[6], *wt_2[6];
    for (int l = 0; l < 6; ++l) {
        wt_q[l] = ws.f16(256 * 256); wt_k[l] = ws.f16(256 * 256);
        wt_v[l] = ws.f16(256 * 256); wt_o[l] = ws.f16(256 * 256);
        wt_1[l] = ws.f16(1024 * 256); wt_2[l] = ws.f16(256 * 1024);
    }
    // agent pointnet
    h16*   objin16 = ws.f16((size_t)R_OBJ * 32);
    float* feat1   = ws.f32((size_t)R_OBJ * 256);
    float* poolObj = ws.f32((size_t)Bn * No * 256);
    h16*   ccObj16 = ws.f16((size_t)R_OBJ * 512);
    h16*   hObj16  = ws.f16((size_t)R_OBJ * 256);
    float* feat3   = ws.f32((size_t)R_OBJ * 256);
    h16*   f3_16   = ws.f16((size_t)R_OBJ * 256);
    h16*   ow0o16  = ws.f16((size_t)R_OBJ * 256);
    float* objout  = ws.f32((size_t)R_OBJ * 256);
    // map pointnet
    h16*   mapin16 = ws.f16((size_t)R_MP * 32);
    h16*   mpA16   = ws.f16((size_t)R_MP * 64);
    h16*   mpB16   = ws.f16((size_t)R_MP * 64);
    float* mpF     = ws.f32((size_t)R_MP * 64);
    float* poolMap = ws.f32((size_t)R_MAP * 64);
    h16*   ccMap16 = ws.f16((size_t)R_MP * 128);
    h16*   pm16    = ws.f16((size_t)R_MAP * 64);
    h16*   mow16   = ws.f16((size_t)R_MAP * 64);
    float* mapout  = ws.f32((size_t)R_MAP * 256);
    // tokens / transformer
    float* X    = ws.f32((size_t)R_TOK * Dm);
    float* PE   = ws.f32((size_t)R_TOK * Dm);
    float* Qb   = ws.f32((size_t)R_TOK * Dm);
    float* Kb   = ws.f32((size_t)R_TOK * Dm);
    float* Vb   = ws.f32((size_t)R_TOK * Dm);
    float* CTX  = ws.f32((size_t)R_TOK * Dm);
    float* GOUT = ws.f32((size_t)R_TOK * Dm);
    float* X1   = ws.f32((size_t)R_TOK * Dm);
    h16*   a16a = ws.f16((size_t)R_TOK * Dm);   // qk16 / ctx16
    h16*   a16b = ws.f16((size_t)R_TOK * Dm);   // x16 / x1_16
    h16*   hid16 = ws.f16((size_t)R_TOK * 1024);
    int*   tmask = ws.i32(R_TOK);
    float* tpos  = ws.f32((size_t)R_TOK * 3);
    int*   idx   = ws.i32((size_t)R_TOK * KNN);
    int*   nbv   = ws.i32((size_t)R_TOK * KNN);
    (void)ws_size;

    hipStream_t st = stream;
    // --- weight conversion ---
    auto cvt = [&](const float* W, h16* Wt, int K, int N, int Kp) {
        k_wt<<<gdiv(N * Kp), 256, 0, st>>>(W, Wt, K, N, Kp);
    };
    cvt(a_pre, wt_apre, 30, 256, 32);
    cvt(a_m0,  wt_am0, 512, 256, 512);
    cvt(a_m1,  wt_am1, 256, 256, 256);
    cvt(a_ow0, wt_aw0, 256, 256, 256);
    cvt(a_ow1, wt_aw1, 256, 256, 256);
    cvt(m_p0,  wt_mp0,  9,  64, 32);
    cvt(m_p1,  wt_mp1, 64,  64, 64);
    cvt(m_p2,  wt_mp2, 64,  64, 64);
    cvt(m_m0,  wt_mm0, 128, 64, 128);
    cvt(m_m1,  wt_mm1, 64,  64, 64);
    cvt(m_ow0, wt_mw0, 64,  64, 64);
    cvt(m_ow1, wt_mw1, 64, 256, 64);
    const float* LY[6][16];
    for (int l = 0; l < 6; ++l) {
        for (int j = 0; j < 16; ++j) LY[l][j] = (const float*)slot[22 + 16 * l + j];
        cvt(LY[l][0],  wt_q[l], 256, 256, 256);
        cvt(LY[l][2],  wt_k[l], 256, 256, 256);
        cvt(LY[l][4],  wt_v[l], 256, 256, 256);
        cvt(LY[l][6],  wt_o[l], 256, 256, 256);
        cvt(LY[l][12], wt_1[l], 256, 1024, 256);
        cvt(LY[l][14], wt_2[l], 1024, 256, 1024);
    }

    // --- agent pointnet (multipoint) ---
    k_obj_in<<<gdiv(R_OBJ * 32), 256, 0, st>>>(obj_trajs, omask, objin16);
    gemm(st, objin16, wt_apre, nullptr, 1.f, 1, feat1, nullptr, R_OBJ, 256, 32);
    k_maskrows<<<gdiv(R_OBJ * 256), 256, 0, st>>>(feat1, omask, R_OBJ, 256);
    k_poolmax<<<gdiv(Bn * No * 256), 256, 0, st>>>(feat1, poolObj, Bn * No, Tt, 256);
    k_concat16<<<gdiv(R_OBJ * 512), 256, 0, st>>>(feat1, poolObj, ccObj16, R_OBJ, 256, 256, Tt);
    gemm(st, ccObj16, wt_am0, nullptr, 1.f, 1, nullptr, hObj16, R_OBJ, 256, 512);
    gemm(st, hObj16,  wt_am1, nullptr, 1.f, 1, feat3, nullptr, R_OBJ, 256, 256);
    k_maskrows<<<gdiv(R_OBJ * 256), 256, 0, st>>>(feat3, omask, R_OBJ, 256);
    k_cast16<<<gdiv(R_OBJ * 256), 256, 0, st>>>(feat3, nullptr, f3_16, R_OBJ * 256);
    gemm(st, f3_16,  wt_aw0, a_ob0, 1.f, 1, nullptr, ow0o16, R_OBJ, 256, 256);
    gemm(st, ow0o16, wt_aw1, a_ob1, 1.f, 0, objout, nullptr, R_OBJ, 256, 256);

    // --- map pointnet ---
    k_map_in<<<gdiv(R_MP * 32), 256, 0, st>>>(map_pl, mapin16);
    gemm(st, mapin16, wt_mp0, nullptr, 1.f, 1, nullptr, mpA16, R_MP, 64, 32);
    gemm(st, mpA16,   wt_mp1, nullptr, 1.f, 1, nullptr, mpB16, R_MP, 64, 64);
    gemm(st, mpB16,   wt_mp2, nullptr, 1.f, 1, mpF, nullptr, R_MP, 64, 64);
    k_maskrows<<<gdiv(R_MP * 64), 256, 0, st>>>(mpF, mmask, R_MP, 64);
    k_poolmax<<<gdiv(R_MAP * 64), 256, 0, st>>>(mpF, poolMap, R_MAP, NPt, 64);
    k_concat16<<<gdiv(R_MP * 128), 256, 0, st>>>(mpF, poolMap, ccMap16, R_MP, 64, 64, NPt);
    gemm(st, ccMap16, wt_mm0, nullptr, 1.f, 1, nullptr, mpA16, R_MP, 64, 128);
    gemm(st, mpA16,   wt_mm1, nullptr, 1.f, 1, mpF, nullptr, R_MP, 64, 64);
    k_maskrows<<<gdiv(R_MP * 64), 256, 0, st>>>(mpF, mmask, R_MP, 64);
    k_poolmax<<<gdiv(R_MAP * 64), 256, 0, st>>>(mpF, poolMap, R_MAP, NPt, 64);
    k_cast16<<<gdiv(R_MAP * 64), 256, 0, st>>>(poolMap, nullptr, pm16, R_MAP * 64);
    gemm(st, pm16,  wt_mw0, m_ob0, 1.f, 1, nullptr, mow16, R_MAP, 64, 64);
    gemm(st, mow16, wt_mw1, m_ob1, 1.f, 0, mapout, nullptr, R_MAP, 256, 64);

    // --- tokens, kNN, positional embedding ---
    k_tok_meta<<<gdiv(R_TOK), 256, 0, st>>>(omask, mmask, opos, mcen, tmask, tpos);
    k_tokens<<<gdiv(R_TOK * Dm), 256, 0, st>>>(objout, mapout, tmask, X);
    k_knn<<<Bn, 256, 0, st>>>(tpos, tmask, idx, nbv);
    k_posemb<<<gdiv(R_TOK * 128), 256, 0, st>>>(tpos, PE);

    // --- 6 local-attention transformer layers ---
    const float qscale = 0.17677669529663687f; // 1/sqrt(32)
    for (int l = 0; l < 6; ++l) {
        const float *bq = LY[l][1], *bk = LY[l][3], *bv = LY[l][5], *bo = LY[l][7];
        const float *g1 = LY[l][8], *be1 = LY[l][9], *g2 = LY[l][10], *be2 = LY[l][11];
        const float *fb1 = LY[l][13], *fb2 = LY[l][15];
        k_cast16<<<gdiv(R_TOK * Dm), 256, 0, st>>>(X, PE, a16a, R_TOK * Dm);      // qk_in
        k_cast16<<<gdiv(R_TOK * Dm), 256, 0, st>>>(X, nullptr, a16b, R_TOK * Dm); // x
        gemm(st, a16a, wt_q[l], bq, qscale, 0, Qb, nullptr, R_TOK, 256, 256);
        gemm(st, a16a, wt_k[l], bk, 1.f,    0, Kb, nullptr, R_TOK, 256, 256);
        gemm(st, a16b, wt_v[l], bv, 1.f,    0, Vb, nullptr, R_TOK, 256, 256);
        k_attn<<<R_TOK, 256, 0, st>>>(Qb, Kb, Vb, idx, nbv, CTX);
        k_cast16<<<gdiv(R_TOK * Dm), 256, 0, st>>>(CTX, nullptr, a16a, R_TOK * Dm);
        gemm(st, a16a, wt_o[l], bo, 1.f, 0, GOUT, nullptr, R_TOK, 256, 256);
        k_ln<<<R_TOK / 8, 256, 0, st>>>(X, GOUT, g1, be1, X1, a16b);   // x1 + f16 copy
        gemm(st, a16b,  wt_1[l], fb1, 1.f, 1, nullptr, hid16, R_TOK, 1024, 256);
        gemm(st, hid16, wt_2[l], fb2, 1.f, 0, GOUT, nullptr, R_TOK, 256, 1024);
        k_ln<<<R_TOK / 8, 256, 0, st>>>(X1, GOUT, g2, be2, X, nullptr); // new x
    }

    // --- final assembly [B, No+P, T, D] ---
    k_out<<<gdiv(out_size), 256, 0, st>>>(X, tmask, (float*)d_out, out_size);
}
